// DeepRMSANetwork_62577673503060
// MI455X (gfx1250) — compile-verified, matrix operand order checked by reading, not run on hardware
//
#include <hip/hip_runtime.h>

typedef __attribute__((ext_vector_type(16))) _Float16 v16h;
typedef __attribute__((ext_vector_type(8)))  float    v8f;
typedef __attribute__((ext_vector_type(4)))  float    v4f;

#define B_   8
#define N_   1024
#define FIN  25
#define H_   8
#define D_   16
#define PSTRIDE 1032   // 16x1024 f16 tile padded: 2064B row stride, 16B aligned, bank-spread

// ---------------------------------------------------------------------------
// Kernel 1: Wh = x @ W[h]  (per b,h), store Wh^T as f16 [bh][d][n] for WMMA-B
// loads; also ei = Wh.a_src, ej = Wh.a_dst, and max_j ej per (b,h).
// One block per (b,h); 256 threads x 4 nodes.
// ---------------------------------------------------------------------------
__global__ __launch_bounds__(256) void gat_prep(
    const float* __restrict__ nf, const float* __restrict__ W,
    const float* __restrict__ a, _Float16* __restrict__ wh16,
    float* __restrict__ ei, float* __restrict__ ej, float* __restrict__ mxej)
{
  __shared__ float Wsm[FIN * D_];
  __shared__ float Asm[2 * D_];
  __shared__ float red[256];
  const int tid = threadIdx.x;
  const int bh  = blockIdx.x;
  const int b   = bh >> 3, h = bh & 7;

  for (int i = tid; i < FIN * D_; i += 256) Wsm[i] = W[h * FIN * D_ + i];
  if (tid < 2 * D_) Asm[tid] = a[h * 2 * D_ + tid];
  __syncthreads();

  float emax = -3.0e38f;
  for (int s = 0; s < 4; ++s) {
    const int n = tid + (s << 8);
    float acc[D_];
#pragma unroll
    for (int d = 0; d < D_; ++d) acc[d] = 0.f;
    const float* xp = nf + (size_t)(b * N_ + n) * FIN;
#pragma unroll
    for (int f = 0; f < FIN; ++f) {
      const float xv = xp[f];
#pragma unroll
      for (int d = 0; d < D_; ++d) acc[d] = fmaf(xv, Wsm[f * D_ + d], acc[d]);
    }
    float se = 0.f, sd = 0.f;
#pragma unroll
    for (int d = 0; d < D_; ++d) {
      se = fmaf(acc[d], Asm[d], se);
      sd = fmaf(acc[d], Asm[D_ + d], sd);
    }
#pragma unroll
    for (int d = 0; d < D_; ++d)
      wh16[((size_t)bh * D_ + d) * N_ + n] = (_Float16)acc[d];
    ei[bh * N_ + n] = se;
    ej[bh * N_ + n] = sd;
    emax = fmaxf(emax, sd);
  }
  red[tid] = emax;
  __syncthreads();
  for (int off = 128; off > 0; off >>= 1) {
    if (tid < off) red[tid] = fmaxf(red[tid], red[tid + off]);
    __syncthreads();
  }
  if (tid == 0) mxej[bh] = red[0];
}

// ---------------------------------------------------------------------------
// Kernel 2: fused masked softmax + alpha write + (alpha @ Wh) via WMMA f16.
// One block per (b, h, 16-row tile); 128 threads = 4 waves.
// Thread t owns columns [8t, 8t+8): all global/LDS traffic is b128.
// Exponent is rank-1: e_ij - (ei_i + mx) = ej_j - mx -> one exp per column.
// Alpha (written once, never re-read) uses non-temporal stores to keep the
// reused adj (32MB x 8 heads) and Wh slabs resident in the 192MB L2.
// ---------------------------------------------------------------------------
__global__ __launch_bounds__(128) void gat_attn(
    const int* __restrict__ adj, const _Float16* __restrict__ wh16,
    const float* __restrict__ ej, const float* __restrict__ mxej,
    float* __restrict__ outH, float* __restrict__ outAlpha)
{
  __shared__ __align__(16) _Float16 p16[16 * PSTRIDE]; // exp tile (f16)
  __shared__ float rowred[16 * 4];                     // per-wave row partials
  __shared__ float inv[16];                            // 1/rowsum
  __shared__ float red[4 * 256];                       // cross-wave C reduce

  const int t    = threadIdx.x;
  const int wave = t >> 5, lane = t & 31;
  const int tile = blockIdx.x, h = blockIdx.y, b = blockIdx.z;
  const int bh   = b * H_ + h;
  const int row0 = tile * 16;
  const int jb   = t << 3;   // 8 owned columns

  const int*      adjB   = adj + ((size_t)(b * N_ + row0)) * N_;
  const float*    ejRow  = ej + (size_t)bh * N_;
  const _Float16* wht    = wh16 + (size_t)bh * D_ * N_;
  float*          alphaB = outAlpha + ((size_t)bh * N_ + row0) * N_;
  const float     mx     = mxej[bh];

  union P8 { _Float16 e[8]; uint4 q; };

  // ---- Stage 0: one exp per owned column (vectorized ej read)
  float ev[8];
  {
    const v4f e0 = *(const v4f*)(ejRow + jb);
    const v4f e1 = *(const v4f*)(ejRow + jb + 4);
    ev[0] = __expf(e0.x - mx); ev[1] = __expf(e0.y - mx);
    ev[2] = __expf(e0.z - mx); ev[3] = __expf(e0.w - mx);
    ev[4] = __expf(e1.x - mx); ev[5] = __expf(e1.y - mx);
    ev[6] = __expf(e1.z - mx); ev[7] = __expf(e1.w - mx);
  }

  // ---- Stage 1: mask, f16 stage to LDS (b128), f32 row sums (wave shuffles)
  for (int r = 0; r < 16; ++r) {
    const int* adjRow = adjB + r * N_;
    const int4 a0 = *(const int4*)(adjRow + jb);
    const int4 a1 = *(const int4*)(adjRow + jb + 4);
    const int m[8] = {a0.x, a0.y, a0.z, a0.w, a1.x, a1.y, a1.z, a1.w};
    P8 p;
    float partial = 0.f;
#pragma unroll
    for (int u = 0; u < 8; ++u) {
      const float pf = (m[u] != 0) ? ev[u] : 0.f;
      p.e[u] = (_Float16)pf;
      partial += (float)p.e[u];
    }
    *(uint4*)&p16[r * PSTRIDE + jb] = p.q;
#pragma unroll
    for (int off = 16; off > 0; off >>= 1)
      partial += __shfl_xor(partial, off);
    if (lane == 0) rowred[(r << 2) + wave] = partial;
  }
  __syncthreads();

  if (t < 16) {
    const float s = rowred[t << 2] + rowred[(t << 2) + 1] +
                    rowred[(t << 2) + 2] + rowred[(t << 2) + 3];
    inv[t] = 1.0f / fmaxf(s, 1e-30f);
  }
  __syncthreads();

  // ---- Stage 2a: write normalized alpha (b128 LDS reads, NT b128 stores)
  for (int r = 0; r < 16; ++r) {
    const float iv = inv[r];
    P8 p;
    p.q = *(const uint4*)&p16[r * PSTRIDE + jb];
    v4f o0, o1;
    o0.x = (float)p.e[0] * iv; o0.y = (float)p.e[1] * iv;
    o0.z = (float)p.e[2] * iv; o0.w = (float)p.e[3] * iv;
    o1.x = (float)p.e[4] * iv; o1.y = (float)p.e[5] * iv;
    o1.z = (float)p.e[6] * iv; o1.w = (float)p.e[7] * iv;
    float* op = alphaB + (size_t)r * N_ + jb;
    __builtin_nontemporal_store(o0, (v4f*)op);
    __builtin_nontemporal_store(o1, (v4f*)(op + 4));
  }

  // ---- Stage 2b: WMMA  C[16x16] += p[16xK] * Wh[KxD], K split over 4 waves
  const int rc = lane & 15;          // A row / B col
  const int kb = (lane >> 4) << 3;   // K sub-offset: 0 (lanes 0-15) / 8 (16-31)

  union HF { v16h h; uint4 q[2]; };
  v8f c = {0.f, 0.f, 0.f, 0.f, 0.f, 0.f, 0.f, 0.f};
  const _Float16* whtC = wht + (size_t)rc * N_;

  for (int kk = 0; kk < 8; ++kk) {
    const int j0 = (wave << 8) + (kk << 5);
    HF fa, fb;
    const _Float16* ap = &p16[rc * PSTRIDE + j0 + kb];
    fa.q[0] = *(const uint4*)ap;         // K = kb .. kb+7
    fa.q[1] = *(const uint4*)(ap + 16);  // K = kb+16 .. kb+23
    const _Float16* bp = whtC + j0 + kb;
    fb.q[0] = *(const uint4*)bp;
    fb.q[1] = *(const uint4*)(bp + 16);
    c = __builtin_amdgcn_wmma_f32_16x16x32_f16(
        /*neg_a=*/false, fa.h, /*neg_b=*/false, fb.h,
        /*c_mod=*/(short)0, c, /*reuse_a=*/false, /*reuse_b=*/false);
  }

#pragma unroll
  for (int r = 0; r < 8; ++r) red[(wave << 8) + (r << 5) + lane] = c[r];
  __syncthreads();

  if (wave == 0) {
#pragma unroll
    for (int r = 0; r < 8; ++r) {
      const int idx = (r << 5) + lane;
      const float s = red[idx] + red[256 + idx] + red[512 + idx] + red[768 + idx];
      const int rowM = r + ((lane >> 4) << 3);
      const float val = fmaxf(s * inv[rowM], 0.f);
      const int node = row0 + rowM;
      float* op = &outH[(size_t)(b * N_ + node) * (H_ * D_) + h * D_ + (lane & 15)];
      __builtin_nontemporal_store(val, op);
    }
  }
}

// ---------------------------------------------------------------------------
extern "C" void kernel_launch(void* const* d_in, const int* in_sizes, int n_in,
                              void* d_out, int out_size, void* d_ws, size_t ws_size,
                              hipStream_t stream) {
  (void)in_sizes; (void)n_in; (void)out_size; (void)ws_size;
  const float* nf  = (const float*)d_in[0];
  const int*   adj = (const int*)d_in[1];
  const float* W   = (const float*)d_in[2];
  const float* a   = (const float*)d_in[3];

  float* out = (float*)d_out;
  float* outH = out;                                        // (8,1024,128) f32
  float* outAlpha = out + (size_t)B_ * N_ * H_ * D_;        // (8,8,1024,1024)

  _Float16* wh16 = (_Float16*)d_ws;                         // 2 MB
  float* fws = (float*)((char*)d_ws + (size_t)B_ * H_ * D_ * N_ * sizeof(_Float16));
  float* ei   = fws;                  // 64*1024
  float* ej   = fws + 65536;          // 64*1024
  float* mxej = fws + 131072;         // 64

  gat_prep<<<B_ * H_, 256, 0, stream>>>(nf, W, a, wh16, ei, ej, mxej);
  gat_attn<<<dim3(N_ / 16, H_, B_), 128, 0, stream>>>(adj, wh16, ej, mxej, outH, outAlpha);
}
